// TextConditionedDynamicLayerAttention_18769007084503
// MI455X (gfx1250) — matrix operand order repaired and checked by reading, not compiled, not run on hardware
//
#include <hip/hip_runtime.h>
#include <math.h>

#define DD  4096
#define TT  512
#define NNx 2048
#define LL  24
#define HH  8
#define HDx 512

typedef unsigned short u16;
typedef __attribute__((ext_vector_type(16))) __bf16 v16bf;
typedef __attribute__((ext_vector_type(8)))  float  v8f;

struct U128 { unsigned int x[4]; };
struct U64s { unsigned int x, y; };
union FragU { U128 q[2]; v16bf v; };

__device__ __forceinline__ u16 f2bf(float f) {
    union { float f; unsigned u; } a; a.f = f;
    unsigned r = a.u + 0x7FFFu + ((a.u >> 16) & 1u);
    return (u16)(r >> 16);
}

// CDNA5 async global->LDS copy (ASYNCcnt). IOFFSET is applied to both the LDS
// and the global address, so one address pair covers a 32B per-lane slice.
__device__ __forceinline__ void async_copy_32B(unsigned lds_addr, const void* gptr) {
    asm volatile("global_load_async_to_lds_b128 %0, %1, off\n\t"
                 "global_load_async_to_lds_b128 %0, %1, off offset:16"
                 :: "v"(lds_addr), "v"(gptr) : "memory");
}
__device__ __forceinline__ void wait_async0() {
    asm volatile("s_wait_asynccnt 0x0" ::: "memory");
}

// ---------------------------------------------------------------- utility
__global__ void zero_kernel(float* p, long long n) {
    long long i = (long long)blockIdx.x * 256 + threadIdx.x;
    if (i < n) p[i] = 0.f;
}

__global__ void pack_bf16_kernel(const float* __restrict__ in, u16* __restrict__ out, long long n) {
    long long i = (long long)blockIdx.x * 256 + threadIdx.x;
    if (i < n) out[i] = f2bf(in[i]);
}

// mean over T rows of text_features -> [D]
__global__ void text_mean_kernel(const float* __restrict__ in, float* __restrict__ out) {
    int d = blockIdx.x * 256 + threadIdx.x;
    float s = 0.f;
    for (int t = 0; t < TT; ++t) s += in[(size_t)t * DD + d];
    out[d] = s * (1.f / TT);
}

// per-layer mean over N (atomic partial sums): grid (D/256, L, N/256)
__global__ void layer_mean_kernel(const float* __restrict__ LF, float* __restrict__ y) {
    int d = blockIdx.x * 256 + threadIdx.x;
    int l = blockIdx.y;
    int nb = blockIdx.z;
    const float* p = LF + ((size_t)l * NNx + (size_t)nb * 256) * DD + d;
    float s = 0.f;
    for (int n = 0; n < 256; ++n) s += p[(size_t)n * DD];
    atomicAdd(&y[(size_t)l * DD + d], s * (1.f / NNx));
}

// row-wise layernorm over D=4096; optional broadcast add vector, optional affine,
// optional bf16 and/or f32 outputs. one block (256 thr) per row.
__global__ __launch_bounds__(256) void ln_rows_kernel(
    const float* __restrict__ in, int ld,
    const float* __restrict__ addv,
    const float* __restrict__ w, const float* __restrict__ b,
    u16* __restrict__ out_bf, float* __restrict__ out_f, int ldo)
{
    __shared__ float sm[256];
    const int row = blockIdx.x, tid = threadIdx.x;
    float x[16];
    float s = 0.f;
#pragma unroll
    for (int j = 0; j < 16; ++j) {
        int c = tid + 256 * j;
        float v = in[(size_t)row * ld + c];
        if (addv) v += addv[c];
        x[j] = v; s += v;
    }
    sm[tid] = s; __syncthreads();
    for (int t = 128; t > 0; t >>= 1) { if (tid < t) sm[tid] += sm[tid + t]; __syncthreads(); }
    float mean = sm[0] * (1.f / DD);
    __syncthreads();
    float ss = 0.f;
#pragma unroll
    for (int j = 0; j < 16; ++j) { float dv = x[j] - mean; ss += dv * dv; }
    sm[tid] = ss; __syncthreads();
    for (int t = 128; t > 0; t >>= 1) { if (tid < t) sm[tid] += sm[tid + t]; __syncthreads(); }
    float inv = rsqrtf(sm[0] * (1.f / DD) + 1e-5f);
#pragma unroll
    for (int j = 0; j < 16; ++j) {
        int c = tid + 256 * j;
        float yv = (x[j] - mean) * inv;
        if (w) yv = yv * w[c] + b[c];
        if (out_bf) out_bf[(size_t)row * ldo + c] = f2bf(yv);
        if (out_f)  out_f [(size_t)row * ldo + c] = yv;
    }
}

// batched "small-M" matvec for the recurrence precompute.
// MODE 0: out[l] += [y_l|e_l](2D) @ Wc           (rows r -> Wc row r)
// MODE 1: out[l] += [y_l|e_l|tg](3D) @ Wz rows   (row = r<D ? r : r+D)
template<int MODE>
__global__ __launch_bounds__(256) void recur_pre_kernel(
    const float* __restrict__ y, const float* __restrict__ e,
    const float* __restrict__ tg, const float* __restrict__ W,
    float* __restrict__ out)
{
    __shared__ float sa[LL * 64];
    const int tid = threadIdx.x;
    const int d = blockIdx.x * 256 + tid;
    const int r0 = blockIdx.y * 256;
    float acc[LL];
#pragma unroll
    for (int l = 0; l < LL; ++l) acc[l] = 0.f;

    for (int sub = 0; sub < 4; ++sub) {
        int rb = r0 + sub * 64;
#pragma unroll
        for (int i = 0; i < 6; ++i) {          // stage 24x64 A values
            int linear = tid + 256 * i;
            int l = linear >> 6, rr = linear & 63;
            int r = rb + rr;
            float v;
            if (MODE == 0) v = (r < DD) ? y[(size_t)l * DD + r] : e[(size_t)l * DD + (r - DD)];
            else v = (r < DD) ? y[(size_t)l * DD + r]
                   : (r < 2 * DD) ? e[(size_t)l * DD + (r - DD)]
                   : tg[r - 2 * DD];
            sa[linear] = v;
        }
        __syncthreads();
        for (int rr = 0; rr < 64; ++rr) {
            int r = rb + rr;
            int wrow = (MODE == 0) ? r : (r < DD ? r : r + DD);
            float wv = W[(size_t)wrow * DD + d];
#pragma unroll
            for (int l = 0; l < LL; ++l) acc[l] += sa[l * 64 + rr] * wv;
        }
        __syncthreads();
    }
#pragma unroll
    for (int l = 0; l < LL; ++l) atomicAdd(&out[(size_t)l * DD + d], acc[l]);
}

__global__ void recur_epi_kernel(float* __restrict__ ct, const float* __restrict__ cb,
                                 float* __restrict__ zb, const float* __restrict__ zbb)
{
    int i = blockIdx.x * 256 + threadIdx.x;   // over L*D
    int dcol = i & (DD - 1);
    ct[i] = tanhf(ct[i] + cb[dcol]);
    zb[i] += zbb[dcol];
}

// per-step GEMV: tmp[d] += sum_r c_prev[r] * Wz[(D+r)*D + d]  (Wz2 stays L2-resident)
__global__ __launch_bounds__(256) void step_gemv_kernel(
    const float* __restrict__ cprev, const float* __restrict__ Wz, float* __restrict__ tmp)
{
    const int d = blockIdx.x * 256 + threadIdx.x;
    const int r0 = blockIdx.y * 256;
    float acc = 0.f;
    for (int rr = 0; rr < 256; ++rr) {
        int r = r0 + rr;
        acc += cprev[r] * Wz[(size_t)(DD + r) * DD + d];
    }
    atomicAdd(&tmp[d], acc);
}

__global__ void step_update_kernel(const float* __restrict__ cprev, const float* __restrict__ zb_l,
                                   const float* __restrict__ ct_l, float* __restrict__ tmp,
                                   float* __restrict__ cnext)
{
    int d = blockIdx.x * 256 + threadIdx.x;
    float zl = (zb_l[d] + tmp[d]) * (1.f / 3.f);
    float z = 1.f / (1.f + expf(-zl));
    float cp = cprev[d];
    cnext[d] = cp + z * (ct_l[d] - cp);
    tmp[d] = 0.f;                              // re-zero for next step
}

// ---------------------------------------------------------------- tiled WMMA GEMM
// C = epi( A(bf16)[M,K] @ B[K,N] + bias ), 128x128 block tile, BK=32, 8 waves (2x4),
// each wave 64x32 = 4x2 wmma tiles. EPI: 0 none, 1 gelu->bf16, 2 tanh+residual.
// A tile staged via async global->LDS (ASYNCcnt); B tile transposed into LDS with
// packed ds_store_b64 (4 bf16 per store).
template<int EPI, int BSRC>
__global__ __launch_bounds__(256) void gemm_wmma_kernel(
    const u16* __restrict__ A, int lda, long long sAz,
    const void* __restrict__ Bv, int ldb, long long sBz,
    const float* __restrict__ bias,
    float* __restrict__ Cf, int ldc,
    u16* __restrict__ Cb, int ldcb,
    long long sCz,
    const float* __restrict__ extra, int lde,
    const float* __restrict__ alpha_p,
    int K)
{
    __shared__ u16 As[128][40];   // 80B row pitch: 16B aligned, conflict-free frag loads
    __shared__ u16 Bs[128][40];   // Bs[n][k] (B transposed into LDS)

    const int tid = threadIdx.x;
    const int z = blockIdx.z;
    const int m0 = blockIdx.y * 128;
    const int n0 = blockIdx.x * 128;

    A += (size_t)z * sAz;
    const float* Bf = (const float*)Bv;
    const u16*   Bh = (const u16*)Bv;
    if (BSRC == 0) Bf += (size_t)z * sBz; else Bh += (size_t)z * sBz;
    const long long cofs = (long long)z * sCz;

    const int wave = tid >> 5, lane = tid & 31;
    const int hi = lane >> 4,  ln = lane & 15;
    const int wm = wave >> 2,  wn = wave & 3;

    v8f acc[4][2];
#pragma unroll
    for (int mi = 0; mi < 4; ++mi)
#pragma unroll
        for (int ni = 0; ni < 2; ++ni)
#pragma unroll
            for (int r = 0; r < 8; ++r) acc[mi][ni][r] = 0.f;

    const int arow = tid >> 1;
    const int acol = (tid & 1) * 16;
    const unsigned ldsA = (unsigned)(uintptr_t)&As[arow][acol];

    for (int k0 = 0; k0 < K; k0 += 32) {
        // A tile: 128x32 bf16, 32B per lane, async straight into LDS
        async_copy_32B(ldsA, A + (size_t)(m0 + arow) * lda + k0 + acol);

        // B tile: each thread covers (n, 4 consecutive k); 4 coalesced global
        // loads -> one packed ds_store_b64 into the transposed tile.
#pragma unroll
        for (int i = 0; i < 4; ++i) {
            int linear = tid + 256 * i;       // 0..1023
            int n  = linear & 127;
            int kg = linear >> 7;             // 0..7
            u16 vv[4];
#pragma unroll
            for (int j = 0; j < 4; ++j) {
                int kc = kg * 4 + j;
                if (BSRC == 0) vv[j] = f2bf(Bf[(size_t)(k0 + kc) * ldb + n0 + n]);
                else           vv[j] = Bh[(size_t)(k0 + kc) * ldb + n0 + n];
            }
            U64s pk;
            pk.x = (unsigned)vv[0] | ((unsigned)vv[1] << 16);
            pk.y = (unsigned)vv[2] | ((unsigned)vv[3] << 16);
            *(U64s*)&Bs[n][kg * 4] = pk;      // 8B aligned (80B pitch)
        }
        wait_async0();
        __syncthreads();

        FragU af[4], bfr[2];
#pragma unroll
        for (int mi = 0; mi < 4; ++mi) {       // ISA A layout: j<8 -> K=hi*8+j ; j>=8 -> K=16+hi*8+(j-8)
            int row = wm * 64 + mi * 16 + ln;
            af[mi].q[0] = *(const U128*)&As[row][hi * 8];
            af[mi].q[1] = *(const U128*)&As[row][16 + hi * 8];
        }
#pragma unroll
        for (int ni = 0; ni < 2; ++ni) {       // ISA B layout: j<8 -> K=hi*8+j ; j>=8 -> K=16+hi*8+(j-8)
            int col = wn * 32 + ni * 16 + ln;
            bfr[ni].q[0] = *(const U128*)&Bs[col][hi * 8];
            bfr[ni].q[1] = *(const U128*)&Bs[col][16 + hi * 8];
        }
#pragma unroll
        for (int mi = 0; mi < 4; ++mi)
#pragma unroll
            for (int ni = 0; ni < 2; ++ni)
                acc[mi][ni] = __builtin_amdgcn_wmma_f32_16x16x32_bf16(
                    false, af[mi].v, false, bfr[ni].v, (short)0, acc[mi][ni], false, false);
        __syncthreads();
    }

    float alphaT = 0.f;
    if (EPI == 2) alphaT = tanhf(alpha_p[0]);

#pragma unroll
    for (int mi = 0; mi < 4; ++mi) {
#pragma unroll
        for (int ni = 0; ni < 2; ++ni) {
            int n = n0 + wn * 32 + ni * 16 + ln;
            float bv = bias ? bias[n] : 0.f;
#pragma unroll
            for (int r = 0; r < 8; ++r) {      // ISA C layout: row = hi*8 + r, col = lane&15
                int m = m0 + wm * 64 + mi * 16 + hi * 8 + r;
                float v = acc[mi][ni][r] + bv;
                float o;
                if (EPI == 0)      o = v;
                else if (EPI == 1) o = 0.5f * v * (1.f + erff(v * 0.70710678118f)); // exact gelu
                else { float t2 = tanhf(v); o = extra[(size_t)m * lde + n] + alphaT * t2; }
                if (Cf) Cf[cofs + (size_t)m * ldc  + n] = o;
                if (Cb) Cb[cofs + (size_t)m * ldcb + n] = f2bf(o);
            }
        }
    }
}

// ---------------------------------------------------------------- attention scores (Q @ K^T, direct-from-global fragments)
__global__ __launch_bounds__(256) void scores_kernel(
    const u16* __restrict__ Q, const u16* __restrict__ Kn, float* __restrict__ S)
{
    const int h = blockIdx.z;
    const int t0 = blockIdx.y * 16;
    const int n0 = blockIdx.x * 128;
    const int tid = threadIdx.x;
    const int wave = tid >> 5, lane = tid & 31;
    const int hi = lane >> 4,  ln = lane & 15;
    const int nw = n0 + wave * 16;

    const u16* qb = Q  + (size_t)(t0 + ln) * DD + h * HDx;
    const u16* kb = Kn + (size_t)(nw + ln) * DD + h * HDx;

    v8f acc;
#pragma unroll
    for (int r = 0; r < 8; ++r) acc[r] = 0.f;

    for (int d = 0; d < HDx; d += 32) {
        FragU a, b;
        a.q[0] = *(const U128*)(qb + d + hi * 16);
        a.q[1] = *(const U128*)(qb + d + hi * 16 + 8);
        b.q[0] = *(const U128*)(kb + d + hi * 16);       // B^T: contiguous along contraction
        b.q[1] = *(const U128*)(kb + d + hi * 16 + 8);
        acc = __builtin_amdgcn_wmma_f32_16x16x32_bf16(
            false, a.v, false, b.v, (short)0, acc, false, false);
    }
    const float scl = 0.044194173824159216f;  // 1/sqrt(512)
#pragma unroll
    for (int r = 0; r < 8; ++r) {
        int t = t0 + hi * 8 + r;
        S[((size_t)h * TT + t) * NNx + nw + ln] = acc[r] * scl;
    }
}

__global__ __launch_bounds__(256) void softmax_kernel(
    const float* __restrict__ S, u16* __restrict__ Wb)
{
    __shared__ float sm[256];
    const size_t row = blockIdx.x;            // h*T + t
    const int tid = threadIdx.x;
    const float* sr = S + row * NNx;
    float x[8];
    float mx = -3.0e38f;
#pragma unroll
    for (int j = 0; j < 8; ++j) { x[j] = sr[tid + 256 * j]; mx = fmaxf(mx, x[j]); }
    sm[tid] = mx; __syncthreads();
    for (int t = 128; t > 0; t >>= 1) { if (tid < t) sm[tid] = fmaxf(sm[tid], sm[tid + t]); __syncthreads(); }
    mx = sm[0]; __syncthreads();
    float sum = 0.f;
#pragma unroll
    for (int j = 0; j < 8; ++j) { x[j] = expf(x[j] - mx); sum += x[j]; }
    sm[tid] = sum; __syncthreads();
    for (int t = 128; t > 0; t >>= 1) { if (tid < t) sm[tid] += sm[tid + t]; __syncthreads(); }
    float inv = 1.f / sm[0];
#pragma unroll
    for (int j = 0; j < 8; ++j) Wb[row * NNx + tid + 256 * j] = f2bf(x[j] * inv);
}

// ---------------------------------------------------------------- host orchestration
extern "C" void kernel_launch(void* const* d_in, const int* in_sizes, int n_in,
                              void* d_out, int out_size, void* d_ws, size_t ws_size,
                              hipStream_t stream)
{
    (void)in_sizes; (void)n_in; (void)out_size; (void)ws_size;
    const float* text  = (const float*)d_in[0];
    const float* layerF= (const float*)d_in[1];
    const float* l_emb = (const float*)d_in[2];
    const float* Wc_w  = (const float*)d_in[3];
    const float* Wc_b  = (const float*)d_in[4];
    const float* Wz_w  = (const float*)d_in[5];
    const float* Wz_b  = (const float*)d_in[6];
    const float* d1_w  = (const float*)d_in[7];
    const float* d1_b  = (const float*)d_in[8];
    const float* d2_w  = (const float*)d_in[9];
    const float* d2_b  = (const float*)d_in[10];
    const float* alpha = (const float*)d_in[11];
    const float* Wq_w  = (const float*)d_in[12];
    const float* Wq_b  = (const float*)d_in[13];
    const float* Wk_w  = (const float*)d_in[14];
    const float* Wk_b  = (const float*)d_in[15];
    const float* Wo_w  = (const float*)d_in[16];
    const float* Wo_b  = (const float*)d_in[17];
    const float* qn_w  = (const float*)d_in[18];
    const float* qn_b  = (const float*)d_in[19];
    const float* kn_w  = (const float*)d_in[20];
    const float* kn_b  = (const float*)d_in[21];
    const float* vn_w  = (const float*)d_in[22];
    const float* vn_b  = (const float*)d_in[23];
    const float* on_w  = (const float*)d_in[24];
    const float* on_b  = (const float*)d_in[25];

    char* base = (char*)d_ws;
    size_t off = 0;
    auto carve = [&](size_t bytes) -> void* {
        void* p = base + off;
        off = (off + bytes + 255) & ~(size_t)255;
        return p;
    };

    float* tg_mean  = (float*)carve((size_t)DD * 4);
    float* tg       = (float*)carve((size_t)DD * 4);
    float* yv       = (float*)carve((size_t)LL * DD * 4);
    float* ct_all   = (float*)carve((size_t)LL * DD * 4);
    float* zb_all   = (float*)carve((size_t)LL * DD * 4);
    float* tmp      = (float*)carve((size_t)DD * 4);
    float* ctx      = (float*)carve((size_t)(LL + 1) * DD * 4);
    float* refreshed= (float*)carve((size_t)NNx * DD * 4);
    float* Pf32     = (float*)carve((size_t)NNx * DD * 4);
    float* Sbuf     = (float*)carve((size_t)HH * TT * NNx * 4);
    float* attnF    = (float*)carve((size_t)TT * DD * 4);
    u16*  Abf       = (u16*)carve((size_t)NNx * DD * 2);
    u16*  Gbf       = (u16*)carve((size_t)NNx * DD * 2);
    u16*  Rbf       = (u16*)carve((size_t)NNx * DD * 2);
    u16*  Tbf       = (u16*)carve((size_t)TT * DD * 2);
    u16*  Qn        = (u16*)carve((size_t)TT * DD * 2);
    u16*  Kn        = (u16*)carve((size_t)NNx * DD * 2);
    u16*  Vn        = (u16*)carve((size_t)NNx * DD * 2);
    u16*  Wbf       = (u16*)carve((size_t)HH * TT * NNx * 2);
    u16*  ATbf      = (u16*)carve((size_t)TT * DD * 2);

    // zero accumulators: yv .. ctx[row0] span (contiguous carve region)
    {
        long long nfl = (long long)(((char*)ctx - (char*)yv) / 4) + DD;
        zero_kernel<<<(unsigned)((nfl + 255) / 256), 256, 0, stream>>>(yv, nfl);
    }

    // text_global = LN(mean(text))
    text_mean_kernel<<<DD / 256, 256, 0, stream>>>(text, tg_mean);
    ln_rows_kernel<<<1, 256, 0, stream>>>(tg_mean, DD, nullptr, nullptr, nullptr, nullptr, tg, DD);

    // per-layer means (one 805MB streaming pass)
    layer_mean_kernel<<<dim3(DD / 256, LL, NNx / 256), 256, 0, stream>>>(layerF, yv);

    // recurrence precompute (weights read exactly once)
    recur_pre_kernel<0><<<dim3(DD / 256, (2 * DD) / 256), 256, 0, stream>>>(yv, l_emb, nullptr, Wc_w, ct_all);
    recur_pre_kernel<1><<<dim3(DD / 256, (3 * DD) / 256), 256, 0, stream>>>(yv, l_emb, tg, Wz_w, zb_all);
    recur_epi_kernel<<<(LL * DD) / 256, 256, 0, stream>>>(ct_all, Wc_b, zb_all, Wz_b);

    // sequential 24-step scan: only c_prev @ Wz2 (64MB, L2-resident) per step
    for (int l = 0; l < LL; ++l) {
        step_gemv_kernel<<<dim3(DD / 256, DD / 256), 256, 0, stream>>>(ctx + (size_t)l * DD, Wz_w, tmp);
        step_update_kernel<<<DD / 256, 256, 0, stream>>>(ctx + (size_t)l * DD, zb_all + (size_t)l * DD,
                                                         ct_all + (size_t)l * DD, tmp, ctx + (size_t)(l + 1) * DD);
    }

    const float* pre23 = layerF + (size_t)(LL - 2) * NNx * DD;   // layer_features[-2]
    const float* c23   = ctx + (size_t)(LL - 1) * DD;            // contexts[L-2]

    // SpatialGate: h = LN(pre23 + c23) -> GEMM1(gelu) -> GEMM2(tanh + residual)
    ln_rows_kernel<<<NNx, 256, 0, stream>>>(pre23, DD, c23, nullptr, nullptr, Abf, nullptr, DD);
    gemm_wmma_kernel<1, 0><<<dim3(DD / 128, NNx / 128, 1), 256, 0, stream>>>(
        Abf, DD, 0, d1_w, DD, 0, d1_b, nullptr, 0, Gbf, DD, 0, nullptr, 0, nullptr, DD);
    gemm_wmma_kernel<2, 0><<<dim3(DD / 128, NNx / 128, 1), 256, 0, stream>>>(
        Gbf, DD, 0, d2_w, DD, 0, d2_b, refreshed, DD, Rbf, DD, 0, pre23, DD, alpha, DD);

    // Q / K / V
    pack_bf16_kernel<<<(TT * DD) / 256, 256, 0, stream>>>(text, Tbf, (long long)TT * DD);
    gemm_wmma_kernel<0, 0><<<dim3(DD / 128, TT / 128, 1), 256, 0, stream>>>(
        Tbf, DD, 0, Wq_w, DD, 0, Wq_b, Pf32, DD, nullptr, 0, 0, nullptr, 0, nullptr, DD);
    ln_rows_kernel<<<TT, 256, 0, stream>>>(Pf32, DD, nullptr, qn_w, qn_b, Qn, nullptr, DD);

    gemm_wmma_kernel<0, 0><<<dim3(DD / 128, NNx / 128, 1), 256, 0, stream>>>(
        Rbf, DD, 0, Wk_w, DD, 0, Wk_b, Pf32, DD, nullptr, 0, 0, nullptr, 0, nullptr, DD);
    ln_rows_kernel<<<NNx, 256, 0, stream>>>(Pf32, DD, nullptr, kn_w, kn_b, Kn, nullptr, DD);
    ln_rows_kernel<<<NNx, 256, 0, stream>>>(refreshed, DD, nullptr, vn_w, vn_b, Vn, nullptr, DD);

    // attention
    scores_kernel<<<dim3(NNx / 128, TT / 16, HH), 256, 0, stream>>>(Qn, Kn, Sbuf);
    softmax_kernel<<<HH * TT, 256, 0, stream>>>(Sbuf, Wbf);
    gemm_wmma_kernel<0, 1><<<dim3(HDx / 128, TT / 128, HH), 256, 0, stream>>>(
        Wbf, NNx, (long long)TT * NNx, Vn, DD, HDx, nullptr, attnF, DD, nullptr, 0, HDx,
        nullptr, 0, nullptr, NNx);

    // output projection + final LN (f32 out)
    pack_bf16_kernel<<<(TT * DD) / 256, 256, 0, stream>>>(attnF, ATbf, (long long)TT * DD);
    gemm_wmma_kernel<0, 0><<<dim3(DD / 128, TT / 128, 1), 256, 0, stream>>>(
        ATbf, DD, 0, Wo_w, DD, 0, Wo_b, Pf32, DD, nullptr, 0, 0, nullptr, 0, nullptr, DD);
    ln_rows_kernel<<<TT, 256, 0, stream>>>(Pf32, DD, nullptr, on_w, on_b, nullptr, (float*)d_out, DD);
}
